// BotDyGNN_3633542332537
// MI455X (gfx1250) — matrix operand order repaired and checked by reading, not compiled, not run on hardware
//
#include <hip/hip_runtime.h>
#include <hip/hip_bf16.h>

typedef __attribute__((ext_vector_type(16))) __bf16 v16bf;
typedef __attribute__((ext_vector_type(8)))  float  v8f;

namespace {
constexpr int T_  = 8;
constexpr int N_  = 10000;
constexpr int E_  = 80000;
constexpr int B_  = 8192;
constexpr int D_  = 128;
constexpr int H_  = 8;
constexpr int DH_ = 16;
constexpr int F_  = 768;
}

// ---------------------------------------------------------------------------
// Weight pre-pack: W[K,Nc] f32 (row-major) -> bf16 fragments in per-lane WMMA
// B-operand order. One thread packs one lane's 16 elements of one fragment.
// Fragment (kt, nt): lane = l15 + 16*hi holds col = nt*16+l15,
// element e -> K index kt*32 + hi*16 + e.   (contiguous k order chosen here, matching the A operand; wave32)
// ---------------------------------------------------------------------------
__global__ void pack_weight_kernel(const float* __restrict__ W, __bf16* __restrict__ P,
                                   int K, int Nc)
{
    const int KT = K >> 5, n16 = Nc >> 4;
    const int total = KT * n16 * 32;
    const int i = blockIdx.x * blockDim.x + threadIdx.x;
    if (i >= total) return;
    const int lane = i & 31;
    const int tile = i >> 5;              // kt * n16 + nt
    const int kt = tile / n16, nt = tile % n16;
    const int l15 = lane & 15, hi = lane >> 4;
    const float* __restrict__ src = W + (size_t)(kt * 32 + hi * 16) * Nc + nt * 16 + l15;
    __bf16* __restrict__ dst = P + (size_t)i * 16;
    #pragma unroll
    for (int e = 0; e < 16; ++e) dst[e] = (__bf16)src[(size_t)e * Nc];
}

// ---------------------------------------------------------------------------
// Fast bf16 WMMA GEMM: C[M,Nc] = act( A[M,K] @ B[K,Nc] + bias + res1 + res2 )
// REQUIRES: M % 16 == 0, Nc % 32 == 0, K % 32 == 0, A rows 16B-aligned,
// B pre-packed by pack_weight_kernel. One wave = 16x32 tile, two
// v_wmma_f32_16x16x32_bf16 per K-step, zero bounds checks in the hot loop.
// Per K-step: 4x b128 (A) + 2x 32B packed (B) + 8 cvt_pk + 2 wmma.
// ---------------------------------------------------------------------------
__global__ __launch_bounds__(256) void gemm_bf16_wmma_kernel(
    const float* __restrict__ A, int lda,
    const __bf16* __restrict__ Bp,        // packed weights
    const float* __restrict__ bias,       // nullable
    const float* __restrict__ res1,       // nullable, ld = ldc
    const float* __restrict__ res2,       // nullable, ld = ldc
    const float* __restrict__ alpha_ptr,  // nullable -> linear, else PReLU slope
    float* __restrict__ C, int ldc,
    int K, int tilesN2, int tiles)
{
    const int wv = (blockIdx.x * blockDim.x + threadIdx.x) >> 5;
    if (wv >= tiles) return;
    const int lane = threadIdx.x & 31;
    const int tm = wv / tilesN2, tn = wv % tilesN2;
    const int row0 = tm << 4, col0 = tn << 5;
    const int l15 = lane & 15;
    const int hi  = lane >> 4;

    // A: lanes 0-15 hold row M=l15 with K run [0..15]; lanes 16-31: [16..31]
    const float* __restrict__ Arow = A + (size_t)(row0 + l15) * lda;

    // Packed B: fragment (kt, nt) at ((kt*n16 + nt)*32 + lane)*16 bf16
    const int n16 = tilesN2 * 2;
    const __bf16* __restrict__ bptr = Bp + ((size_t)(2 * tn) * 32 + lane) * 16;
    const size_t bstep = (size_t)n16 * 32 * 16;

    v8f acc0 = {0.f, 0.f, 0.f, 0.f, 0.f, 0.f, 0.f, 0.f};
    v8f acc1 = {0.f, 0.f, 0.f, 0.f, 0.f, 0.f, 0.f, 0.f};

    for (int k0 = 0; k0 < K; k0 += 32) {
        __builtin_prefetch(Arow + k0 + 64, 0, 0);   // global_prefetch_b8 (speculative)

        const float4 a0 = *(const float4*)(Arow + k0 + hi * 16);
        const float4 a1 = *(const float4*)(Arow + k0 + hi * 16 + 4);
        const float4 a2 = *(const float4*)(Arow + k0 + hi * 16 + 8);
        const float4 a3 = *(const float4*)(Arow + k0 + hi * 16 + 12);
        v16bf a;
        a[0]  = (__bf16)a0.x; a[1]  = (__bf16)a0.y; a[2]  = (__bf16)a0.z; a[3]  = (__bf16)a0.w;
        a[4]  = (__bf16)a1.x; a[5]  = (__bf16)a1.y; a[6]  = (__bf16)a1.z; a[7]  = (__bf16)a1.w;
        a[8]  = (__bf16)a2.x; a[9]  = (__bf16)a2.y; a[10] = (__bf16)a2.z; a[11] = (__bf16)a2.w;
        a[12] = (__bf16)a3.x; a[13] = (__bf16)a3.y; a[14] = (__bf16)a3.z; a[15] = (__bf16)a3.w;

        const v16bf b0 = *(const v16bf*)(bptr);          // cols col0..col0+15
        const v16bf b1 = *(const v16bf*)(bptr + 512);    // cols col0+16..col0+31

        acc0 = __builtin_amdgcn_wmma_f32_16x16x32_bf16(false, a, false, b0, (short)0, acc0, false, false);
        acc1 = __builtin_amdgcn_wmma_f32_16x16x32_bf16(false, a, false, b1, (short)0, acc1, false, false);

        bptr += bstep;
    }

    const int bc0 = col0 + l15;
    const int bc1 = bc0 + 16;
    const float bv0 = bias ? bias[bc0] : 0.0f;
    const float bv1 = bias ? bias[bc1] : 0.0f;
    const float al  = alpha_ptr ? alpha_ptr[0] : 0.0f;
    // C/D layout: lane 0-15 -> N=l15, VGPR r = row r; lanes 16-31 -> row 8+r
    #pragma unroll
    for (int r = 0; r < 8; ++r) {
        const int crow = row0 + ((lane < 16) ? r : (8 + r));
        const size_t i0 = (size_t)crow * ldc + bc0;
        const size_t i1 = (size_t)crow * ldc + bc1;
        float v0 = acc0[r] + bv0;
        float v1 = acc1[r] + bv1;
        if (res1) { v0 += res1[i0]; v1 += res1[i1]; }
        if (res2) { v0 += res2[i0]; v1 += res2[i1]; }
        if (alpha_ptr) {
            v0 = (v0 >= 0.0f) ? v0 : al * v0;
            v1 = (v1 >= 0.0f) ? v1 : al * v1;
        }
        C[i0] = v0;
        C[i1] = v1;
    }
}

// Tiny-K encode GEMM (K = 5 or 3, Nc = 32): one thread per output element.
__global__ void small_gemm_kernel(const float* __restrict__ A, int lda, int K,
                                  const float* __restrict__ Bm,   // [K,32]
                                  const float* __restrict__ bias,
                                  const float* __restrict__ alpha_ptr,
                                  float* __restrict__ C, int ldc, int total)
{
    const int i = blockIdx.x * blockDim.x + threadIdx.x;
    if (i >= total) return;
    const int row = i >> 5, col = i & 31;
    const float* __restrict__ ar = A + (size_t)row * lda;
    float acc = bias[col];
    for (int k2 = 0; k2 < K; ++k2) acc += ar[k2] * Bm[k2 * 32 + col];
    const float al = alpha_ptr[0];
    C[(size_t)row * ldc + col] = (acc >= 0.0f) ? acc : al * acc;
}

// Classifier (Nc = 2, K = 128): one thread per output element.
__global__ void classifier_kernel(const float* __restrict__ A, const float* __restrict__ W,
                                  const float* __restrict__ bias, float* __restrict__ out, int rows)
{
    const int i = blockIdx.x * blockDim.x + threadIdx.x;
    if (i >= rows * 2) return;
    const int row = i >> 1, c = i & 1;
    const float* __restrict__ ar = A + (size_t)row * D_;
    float acc = bias[c];
    #pragma unroll 8
    for (int k2 = 0; k2 < D_; ++k2) acc += ar[k2] * W[k2 * 2 + c];
    out[i] = acc;
}

// ---------------------------------------------------------------------------
// Graph attention (TransformerConv) segment kernels
// ---------------------------------------------------------------------------
__global__ void init_graph_kernel(float* __restrict__ m, float* __restrict__ den,
                                  float* __restrict__ agg, int NH, int ND)
{
    const int i = blockIdx.x * blockDim.x + threadIdx.x;
    if (i < NH) { ((int*)m)[i] = 0xFF800000; den[i] = 0.0f; }  // -inf
    if (i < ND) agg[i] = 0.0f;
}

__global__ void edge_score_max_kernel(const int* __restrict__ src, const int* __restrict__ dst,
                                      const float* __restrict__ q, const float* __restrict__ k,
                                      float* __restrict__ s, float* __restrict__ m, int Eh)
{
    const int i = blockIdx.x * blockDim.x + threadIdx.x;
    if (i >= Eh) return;
    const int e = i >> 3, h = i & 7;
    const int sN = src[e], dN = dst[e];
    const float* __restrict__ qp = q + (size_t)dN * D_ + h * DH_;
    const float* __restrict__ kp = k + (size_t)sN * D_ + h * DH_;
    float acc = 0.f;
    #pragma unroll
    for (int d = 0; d < DH_; ++d) acc += qp[d] * kp[d];
    acc *= 0.25f;                        // 1/sqrt(16)
    s[i] = acc;
    float* addr = m + (size_t)dN * H_ + h;
    if (acc >= 0.f) atomicMax((int*)addr, __float_as_int(acc));
    else            atomicMin((unsigned int*)addr, __float_as_uint(acc));
}

__global__ void edge_exp_sum_kernel(const int* __restrict__ dst, float* __restrict__ s,
                                    const float* __restrict__ m, float* __restrict__ den, int Eh)
{
    const int i = blockIdx.x * blockDim.x + threadIdx.x;
    if (i >= Eh) return;
    const int e = i >> 3, h = i & 7;
    const int dN = dst[e];
    const float ev = __expf(s[i] - m[(size_t)dN * H_ + h]);
    s[i] = ev;
    atomicAdd(den + (size_t)dN * H_ + h, ev);
}

__global__ void edge_scatter_kernel(const int* __restrict__ src, const int* __restrict__ dst,
                                    const float* __restrict__ s, const float* __restrict__ den,
                                    const float* __restrict__ v, float* __restrict__ agg, int Eh)
{
    const int i = blockIdx.x * blockDim.x + threadIdx.x;
    if (i >= Eh) return;
    const int e = i >> 3, h = i & 7;
    const int sN = src[e], dN = dst[e];
    const float alpha = s[i] / den[(size_t)dN * H_ + h];
    const float* __restrict__ vp = v + (size_t)sN * D_ + h * DH_;
    float* __restrict__ ap = agg + (size_t)dN * D_ + h * DH_;
    #pragma unroll
    for (int d = 0; d < DH_; ++d) atomicAdd(ap + d, alpha * vp[d]);
}

// Take first B rows of conv2 output -> so[b,t,:]
__global__ void copy_so_kernel(const float* __restrict__ o, float* __restrict__ so, int t)
{
    const int i = blockIdx.x * blockDim.x + threadIdx.x;
    if (i >= B_ * D_) return;
    const int b = i >> 7, d = i & 127;
    so[((size_t)b * T_ + t) * D_ + d] = o[(size_t)b * D_ + d];
}

// ---------------------------------------------------------------------------
// Temporal: fused embed-add + LayerNorm (one wave32 per row, 4 elems/lane)
// ---------------------------------------------------------------------------
__global__ __launch_bounds__(256) void ti_ln_kernel(
    const float* __restrict__ so, const float* __restrict__ cc, const float* __restrict__ blr,
    const float* __restrict__ Wcc, const float* __restrict__ bcc,
    const float* __restrict__ Wbl, const float* __restrict__ bbl,
    const float* __restrict__ pos, const float* __restrict__ g, const float* __restrict__ be,
    float* __restrict__ out, int rows)
{
    const int wv = (blockIdx.x * blockDim.x + threadIdx.x) >> 5;
    const int lane = threadIdx.x & 31;
    if (wv >= rows) return;
    const int t = wv & (T_ - 1);
    const float ccv = cc[wv];
    const float blv = blr[wv];
    const float* __restrict__ srow = so + (size_t)wv * D_;
    float vals[4];
    float sum = 0.f;
    #pragma unroll
    for (int j = 0; j < 4; ++j) {
        const int d = lane + j * 32;
        float v = srow[d] + pos[t * D_ + d] + ccv * Wcc[d] + bcc[d] + blv * Wbl[d] + bbl[d];
        vals[j] = v; sum += v;
    }
    #pragma unroll
    for (int off = 16; off > 0; off >>= 1) sum += __shfl_xor(sum, off, 32);
    const float mean = sum * (1.0f / D_);
    float var = 0.f;
    #pragma unroll
    for (int j = 0; j < 4; ++j) { const float dv = vals[j] - mean; var += dv * dv; }
    #pragma unroll
    for (int off = 16; off > 0; off >>= 1) var += __shfl_xor(var, off, 32);
    const float inv = rsqrtf(var * (1.0f / D_) + 1e-5f);
    float* __restrict__ orow = out + (size_t)wv * D_;
    #pragma unroll
    for (int j = 0; j < 4; ++j) {
        const int d = lane + j * 32;
        orow[d] = (vals[j] - mean) * inv * g[d] + be[d];
    }
}

__global__ __launch_bounds__(256) void add_ln_kernel(
    const float* __restrict__ x, const float* __restrict__ y,
    const float* __restrict__ g, const float* __restrict__ be,
    float* __restrict__ out, int rows)
{
    const int wv = (blockIdx.x * blockDim.x + threadIdx.x) >> 5;
    const int lane = threadIdx.x & 31;
    if (wv >= rows) return;
    const float* __restrict__ xr = x + (size_t)wv * D_;
    const float* __restrict__ yr = y + (size_t)wv * D_;
    float vals[4];
    float sum = 0.f;
    #pragma unroll
    for (int j = 0; j < 4; ++j) {
        const int d = lane + j * 32;
        const float v = xr[d] + yr[d];
        vals[j] = v; sum += v;
    }
    #pragma unroll
    for (int off = 16; off > 0; off >>= 1) sum += __shfl_xor(sum, off, 32);
    const float mean = sum * (1.0f / D_);
    float var = 0.f;
    #pragma unroll
    for (int j = 0; j < 4; ++j) { const float dv = vals[j] - mean; var += dv * dv; }
    #pragma unroll
    for (int off = 16; off > 0; off >>= 1) var += __shfl_xor(var, off, 32);
    const float inv = rsqrtf(var * (1.0f / D_) + 1e-5f);
    float* __restrict__ orow = out + (size_t)wv * D_;
    #pragma unroll
    for (int j = 0; j < 4; ++j) {
        const int d = lane + j * 32;
        orow[d] = (vals[j] - mean) * inv * g[d] + be[d];
    }
}

// Causal temporal attention: one thread per (b,h,t); T=8, DH=16 in registers.
__global__ void temporal_attn_kernel(const float* __restrict__ q, const float* __restrict__ k,
                                     const float* __restrict__ v, float* __restrict__ out, int total)
{
    const int i = blockIdx.x * blockDim.x + threadIdx.x;
    if (i >= total) return;
    const int t = i & (T_ - 1);
    const int h = (i >> 3) & (H_ - 1);
    const int b = i >> 6;
    const float* __restrict__ qp = q + ((size_t)b * T_ + t) * D_ + h * DH_;
    float sc[T_];
    float mx = -3.4e38f;
    for (int s2 = 0; s2 <= t; ++s2) {
        const float* __restrict__ kp = k + ((size_t)b * T_ + s2) * D_ + h * DH_;
        float a = 0.f;
        #pragma unroll
        for (int d = 0; d < DH_; ++d) a += qp[d] * kp[d];
        a *= 0.25f;
        sc[s2] = a;
        mx = fmaxf(mx, a);
    }
    float den = 0.f;
    for (int s2 = 0; s2 <= t; ++s2) { sc[s2] = __expf(sc[s2] - mx); den += sc[s2]; }
    const float rden = 1.0f / den;
    float o[DH_];
    #pragma unroll
    for (int d = 0; d < DH_; ++d) o[d] = 0.f;
    for (int s2 = 0; s2 <= t; ++s2) {
        const float al = sc[s2] * rden;
        const float* __restrict__ vp = v + ((size_t)b * T_ + s2) * D_ + h * DH_;
        #pragma unroll
        for (int d = 0; d < DH_; ++d) o[d] += al * vp[d];
    }
    float* __restrict__ op = out + ((size_t)b * T_ + t) * D_ + h * DH_;
    #pragma unroll
    for (int d = 0; d < DH_; ++d) op[d] = o[d];
}

// ---------------------------------------------------------------------------
// Host-side helpers
// ---------------------------------------------------------------------------
static void launch_gemm(const float* A, int lda, const __bf16* Bp,
                        const float* bias, const float* res1, const float* res2,
                        const float* alpha, float* C, int ldc,
                        int M, int Nc, int K, hipStream_t stream)
{
    const int tilesN2 = Nc / 32;             // Nc % 32 == 0 guaranteed by call sites
    const int tiles   = (M / 16) * tilesN2;  // M % 16 == 0 guaranteed by call sites
    const int blocks  = (tiles + 7) / 8;     // 8 waves / 256-thread block
    gemm_bf16_wmma_kernel<<<blocks, 256, 0, stream>>>(
        A, lda, Bp, bias, res1, res2, alpha, C, ldc, K, tilesN2, tiles);
}

static void launch_graph_attention(const int* src, const int* dst,
                                   const float* q, const float* k, const float* v,
                                   float* s, float* m, float* den, float* agg,
                                   hipStream_t stream)
{
    const int Eh = E_ * H_;
    init_graph_kernel<<<(N_ * D_ + 255) / 256, 256, 0, stream>>>(m, den, agg, N_ * H_, N_ * D_);
    edge_score_max_kernel<<<(Eh + 255) / 256, 256, 0, stream>>>(src, dst, q, k, s, m, Eh);
    edge_exp_sum_kernel<<<(Eh + 255) / 256, 256, 0, stream>>>(dst, s, m, den, Eh);
    edge_scatter_kernel<<<(Eh + 255) / 256, 256, 0, stream>>>(src, dst, s, den, v, agg, Eh);
}

extern "C" void kernel_launch(void* const* d_in, const int* in_sizes, int n_in,
                              void* d_out, int out_size, void* d_ws, size_t ws_size,
                              hipStream_t stream)
{
    const float* des   = (const float*)d_in[0];
    const float* tweet = (const float*)d_in[1];
    const float* nump  = (const float*)d_in[2];
    const float* catp  = (const float*)d_in[3];
    const int*   ei    = (const int*)d_in[4];
    const float* cc    = (const float*)d_in[5];
    const float* blr   = (const float*)d_in[6];
    // d_in[7] exist_nodes, d_in[8] current_batch_size: unused
    const float* Wn = (const float*)d_in[9];   const float* bn = (const float*)d_in[10];
    const float* Wc = (const float*)d_in[11];  const float* bc = (const float*)d_in[12];
    const float* Wd = (const float*)d_in[13];  const float* bd = (const float*)d_in[14];
    const float* Wt = (const float*)d_in[15];  const float* bt = (const float*)d_in[16];
    const float* Wtot = (const float*)d_in[17]; const float* btot = (const float*)d_in[18];
    const float* a_feat = (const float*)d_in[19];
    const float* c1[8]; for (int i = 0; i < 8; ++i) c1[i] = (const float*)d_in[20 + i];
    const float* c2[8]; for (int i = 0; i < 8; ++i) c2[i] = (const float*)d_in[28 + i];
    const float* a_struct = (const float*)d_in[36];
    const float* Wcc = (const float*)d_in[37], *bcc = (const float*)d_in[38];
    const float* Wbl = (const float*)d_in[39], *bbl = (const float*)d_in[40];
    const float* WqT = (const float*)d_in[41], *WkT = (const float*)d_in[42], *WvT = (const float*)d_in[43];
    const float* lng = (const float*)d_in[44], *lnb = (const float*)d_in[45];
    const float* Wf1 = (const float*)d_in[46], *bf1 = (const float*)d_in[47];
    const float* Wf2 = (const float*)d_in[48], *bf2 = (const float*)d_in[49];
    const float* a_temp = (const float*)d_in[50];
    const float* pos = (const float*)d_in[51];

    float* ws = (float*)d_ws;
    const size_t S = (size_t)B_ * T_ * D_;       // 8.39M floats per [B,T,D] buffer
    float* so = ws;
    float* ti = ws + S;
    float* qT = ws + 2 * S;
    float* kT = ws + 3 * S;
    float* vT = ws + 4 * S;
    float* ao = ws + 5 * S;

    // Graph-phase scratch overlays qT/kT (dead until temporal qkv GEMMs)
    const size_t ND = (size_t)N_ * D_;
    float* gx   = qT;            // x
    float* gq   = gx + ND;       // q
    float* gk   = gq + ND;       // k
    float* gv   = gk + ND;       // v
    float* ga   = gv + ND;       // agg
    float* go   = ga + ND;       // h buffer / conv1 output
    float* gs   = go + ND;       // per-edge scores [E,H]
    float* gm   = gs + (size_t)E_ * H_;   // segment max [N,H]
    float* gden = gm + (size_t)N_ * H_;   // segment sum [N,H]

    // ---------------- pre-pack all WMMA weights to bf16 fragments ----------
    __bf16* pw_cursor = (__bf16*)(ws + 6 * S);
    auto pack = [&](const float* W, int K, int Nc) -> __bf16* {
        __bf16* p = pw_cursor;
        const int total = (K / 32) * (Nc / 16) * 32;
        pw_cursor += (size_t)K * Nc;      // total*16 == K*Nc
        pack_weight_kernel<<<(total + 255) / 256, 256, 0, stream>>>(W, p, K, Nc);
        return p;
    };
    __bf16* pWd   = pack(Wd,   F_, 32);
    __bf16* pWt   = pack(Wt,   F_, 32);
    __bf16* pWtot = pack(Wtot, D_, D_);
    __bf16* pc1q  = pack(c1[0], D_, D_);
    __bf16* pc1k  = pack(c1[2], D_, D_);
    __bf16* pc1v  = pack(c1[4], D_, D_);
    __bf16* pc1s  = pack(c1[6], D_, D_);
    __bf16* pc2q  = pack(c2[0], D_, D_);
    __bf16* pc2k  = pack(c2[2], D_, D_);
    __bf16* pc2v  = pack(c2[4], D_, D_);
    __bf16* pc2s  = pack(c2[6], D_, D_);
    __bf16* pWqT  = pack(WqT, D_, D_);
    __bf16* pWkT  = pack(WkT, D_, D_);
    __bf16* pWvT  = pack(WvT, D_, D_);
    __bf16* pWf1  = pack(Wf1, D_, D_);

    // ---------------- per-snapshot graph pipeline ----------------
    for (int t = 0; t < T_; ++t) {
        const float* des_t = des   + (size_t)t * N_ * F_;
        const float* tw_t  = tweet + (size_t)t * N_ * F_;
        const float* np_t  = nump  + (size_t)t * N_ * 5;
        const float* cp_t  = catp  + (size_t)t * N_ * 3;
        const int* src = ei + (size_t)t * 2 * E_;
        const int* dst = src + E_;

        // feature encode -> h (go), 4 column blocks of 32
        small_gemm_kernel<<<(N_ * 32 + 255) / 256, 256, 0, stream>>>(
            np_t, 5, 5, Wn, bn, a_feat, go + 0, D_, N_ * 32);
        small_gemm_kernel<<<(N_ * 32 + 255) / 256, 256, 0, stream>>>(
            cp_t, 3, 3, Wc, bc, a_feat, go + 32, D_, N_ * 32);
        launch_gemm(des_t, F_, pWd, bd, nullptr, nullptr, a_feat, go + 64, D_, N_, 32, F_, stream);
        launch_gemm(tw_t,  F_, pWt, bt, nullptr, nullptr, a_feat, go + 96, D_, N_, 32, F_, stream);
        // x = prelu(h @ Wtot + btot)
        launch_gemm(go, D_, pWtot, btot, nullptr, nullptr, a_feat, gx, D_, N_, D_, D_, stream);

        // conv1: q/k/v, segment softmax+scatter, skip GEMM fused with agg residual + PReLU
        launch_gemm(gx, D_, pc1q, c1[1], nullptr, nullptr, nullptr, gq, D_, N_, D_, D_, stream);
        launch_gemm(gx, D_, pc1k, c1[3], nullptr, nullptr, nullptr, gk, D_, N_, D_, D_, stream);
        launch_gemm(gx, D_, pc1v, c1[5], nullptr, nullptr, nullptr, gv, D_, N_, D_, D_, stream);
        launch_graph_attention(src, dst, gq, gk, gv, gs, gm, gden, ga, stream);
        launch_gemm(gx, D_, pc1s, c1[7], ga, nullptr, a_struct, go, D_, N_, D_, D_, stream);

        // conv2 (input go), with extra +x residual, output -> gx (per-element safe)
        launch_gemm(go, D_, pc2q, c2[1], nullptr, nullptr, nullptr, gq, D_, N_, D_, D_, stream);
        launch_gemm(go, D_, pc2k, c2[3], nullptr, nullptr, nullptr, gk, D_, N_, D_, D_, stream);
        launch_gemm(go, D_, pc2v, c2[5], nullptr, nullptr, nullptr, gv, D_, N_, D_, D_, stream);
        launch_graph_attention(src, dst, gq, gk, gv, gs, gm, gden, ga, stream);
        launch_gemm(go, D_, pc2s, c2[7], ga, gx, a_struct, gx, D_, N_, D_, D_, stream);

        copy_so_kernel<<<(B_ * D_ + 255) / 256, 256, 0, stream>>>(gx, so, t);
    }

    // ---------------- temporal pipeline ----------------
    const int rows = B_ * T_;
    ti_ln_kernel<<<(rows + 7) / 8, 256, 0, stream>>>(so, cc, blr, Wcc, bcc, Wbl, bbl,
                                                     pos, lng, lnb, ti, rows);
    launch_gemm(ti, D_, pWqT, nullptr, nullptr, nullptr, nullptr, qT, D_, rows, D_, D_, stream);
    launch_gemm(ti, D_, pWkT, nullptr, nullptr, nullptr, nullptr, kT, D_, rows, D_, D_, stream);
    launch_gemm(ti, D_, pWvT, nullptr, nullptr, nullptr, nullptr, vT, D_, rows, D_, D_, stream);
    temporal_attn_kernel<<<(B_ * H_ * T_ + 255) / 256, 256, 0, stream>>>(qT, kT, vT, ao, B_ * H_ * T_);
    add_ln_kernel<<<(rows + 7) / 8, 256, 0, stream>>>(ao, so, lng, lnb, ti, rows);
    float* hid = kT;  // reuse
    launch_gemm(ti, D_, pWf1, bf1, nullptr, nullptr, a_temp, hid, D_, rows, D_, D_, stream);
    classifier_kernel<<<(rows * 2 + 255) / 256, 256, 0, stream>>>(hid, Wf2, bf2, (float*)d_out, rows);
}
